// Head_18004502905486
// MI455X (gfx1250) — matrix operand order repaired and checked by reading, not compiled, not run on hardware
//
#include <hip/hip_runtime.h>

#define B_  8
#define T_  2048
#define C_  1024
#define DK_ 64

typedef __attribute__((ext_vector_type(16))) __bf16 v16bf;
typedef __attribute__((ext_vector_type(8)))  __bf16 v8bf;
typedef __attribute__((ext_vector_type(8)))  float  v8f;

// Build a 16-element bf16 A/B fragment from two contiguous 8-element (16B) chunks.
static __device__ __forceinline__ v16bf ld_frag(const __bf16* p0, const __bf16* p1) {
  v8bf lo = *(const v8bf*)p0;
  v8bf hi = *(const v8bf*)p1;
  v16bf r;
#pragma unroll
  for (int i = 0; i < 8; ++i) { r[i] = lo[i]; r[i + 8] = hi[i]; }
  return r;
}

static __device__ __forceinline__ v8f wmma_bf16(v16bf a, v16bf b, v8f c) {
  // (neg_a, A, neg_b, B, c_mod, C, reuse_a, reuse_b)
  return __builtin_amdgcn_wmma_f32_16x16x32_bf16(false, a, false, b, (short)0, c,
                                                 false, false);
}

// ---------------------------------------------------------------------------
// Kernel 0: transpose + convert weights once: Wt[w][n][k] = bf16(W_w[k][n]).
// 768 KB read once; makes every WMMA B-fragment a contiguous 32B load.
// ---------------------------------------------------------------------------
__global__ __launch_bounds__(256) void wtrans_kernel(
    const float* __restrict__ Wq, const float* __restrict__ Wk,
    const float* __restrict__ Wv, __bf16* __restrict__ Wt) {
  const int idx = blockIdx.x * 256 + threadIdx.x;  // 3*64*1024 threads
  const int k = idx & (C_ - 1);
  const int n = (idx >> 10) & (DK_ - 1);
  const int w = idx >> 16;
  const float* W = (w == 0) ? Wq : (w == 1) ? Wk : Wv;
  Wt[idx] = (__bf16)W[k * DK_ + n];
}

// ---------------------------------------------------------------------------
// Kernel 1: QKV projection.  Each wave computes 16 rows x 64 cols of Q, K, V.
// Q gets bias + 1/sqrt(dk) folded in.  V is stored transposed [b][d][t] so the
// P*V B-fragments in the attention kernel are contiguous loads.
// ---------------------------------------------------------------------------
__global__ __launch_bounds__(128) void qkv_proj_kernel(
    const float* __restrict__ X, const __bf16* __restrict__ Wt,
    const float* __restrict__ bq, const float* __restrict__ bk,
    const float* __restrict__ bv,
    __bf16* __restrict__ Qw, __bf16* __restrict__ Kw, __bf16* __restrict__ Vt) {
  const int lane = threadIdx.x & 31;
  const int wave = threadIdx.x >> 5;
  const int half = lane >> 4;
  const int l16  = lane & 15;
  const int m0   = (blockIdx.x * 4 + wave) * 16;  // global output row base

  v8f zero = {};
  v8f accq[4], acck[4], accv[4];
#pragma unroll
  for (int i = 0; i < 4; ++i) { accq[i] = zero; acck[i] = zero; accv[i] = zero; }

  const float* xr = X + (size_t)(m0 + l16) * C_;  // A-frag: lane = row M
  for (int dc = 0; dc < C_; dc += 32) {
    // A fragment 16x32: lane holds K = dc+16*half..+15 (contiguous)
    v8f f0 = *(const v8f*)(xr + dc + 16 * half);
    v8f f1 = *(const v8f*)(xr + dc + 16 * half + 8);
    v16bf a;
#pragma unroll
    for (int i = 0; i < 8; ++i) { a[i] = (__bf16)f0[i]; a[i + 8] = (__bf16)f1[i]; }

    const int kb = dc + 16 * half;  // B-frag: lane holds 16 contiguous K at col N
#pragma unroll
    for (int nt = 0; nt < 4; ++nt) {
      const int n = nt * 16 + l16;
      const __bf16* pq = Wt + ((size_t)(0 * DK_ + n) << 10) + kb;
      const __bf16* pk = Wt + ((size_t)(1 * DK_ + n) << 10) + kb;
      const __bf16* pv = Wt + ((size_t)(2 * DK_ + n) << 10) + kb;
      accq[nt] = wmma_bf16(a, ld_frag(pq, pq + 8), accq[nt]);
      acck[nt] = wmma_bf16(a, ld_frag(pk, pk + 8), acck[nt]);
      accv[nt] = wmma_bf16(a, ld_frag(pv, pv + 8), accv[nt]);
    }
  }

  const int b  = m0 / T_;
  const int t0 = m0 % T_;
#pragma unroll
  for (int nt = 0; nt < 4; ++nt) {
    const int n = nt * 16 + l16;
    const float bqv = bq[n], bkv = bk[n], bvv = bv[n];
#pragma unroll
    for (int r = 0; r < 8; ++r) {
      const int row = m0 + r + 8 * half;  // C-layout: M = r + 8*half, N = l16
      const int tt  = t0 + r + 8 * half;
      Qw[(size_t)row * DK_ + n] = (__bf16)((accq[nt][r] + bqv) * 0.125f);
      Kw[(size_t)row * DK_ + n] = (__bf16)(acck[nt][r] + bkv);
      Vt[((size_t)b * DK_ + n) * T_ + tt] = (__bf16)(accv[nt][r] + bvv);
    }
  }
}

// ---------------------------------------------------------------------------
// Kernel 2: causal flash attention, 64-key blocks.  4 waves/block; wave owns
// 16 query rows.  One online-softmax update (max/sum butterfly + O rescale)
// amortized over 64 keys; P relayout C->A via per-wave LDS tile.
// ---------------------------------------------------------------------------
__global__ __launch_bounds__(128) void attn_kernel(
    const __bf16* __restrict__ Qw, const __bf16* __restrict__ Kw,
    const __bf16* __restrict__ Vt, float* __restrict__ out) {
  const int lane = threadIdx.x & 31;
  const int wave = threadIdx.x >> 5;
  const int half = lane >> 4;
  const int l16  = lane & 15;
  const int b    = blockIdx.y;
  const int q0   = blockIdx.x * 64 + wave * 16;

  __shared__ __align__(16) __bf16 pbuf[4][16][64];

  const __bf16* Qb = Qw + (size_t)b * T_ * DK_;
  const __bf16* Kb = Kw + (size_t)b * T_ * DK_;
  const __bf16* Vb = Vt + (size_t)b * DK_ * T_;

  // Q A-fragments for K-dim 64 (two 16x32 fragments), resident for whole loop.
  const __bf16* qrow = Qb + (size_t)(q0 + l16) * DK_;
  v16bf qa0 = ld_frag(qrow + 16 * half, qrow + 16 * half + 8);
  v16bf qa1 = ld_frag(qrow + 32 + 16 * half, qrow + 40 + 16 * half);

  v8f zero = {};
  v8f o[4];
#pragma unroll
  for (int i = 0; i < 4; ++i) o[i] = zero;
  float m[8], l[8];
#pragma unroll
  for (int r = 0; r < 8; ++r) { m[r] = -1e30f; l[r] = 0.0f; }

  const int smax = q0 + 15;
  for (int s0 = 0; s0 <= smax; s0 += 64) {
    // --- S = Q * K^T for 64 keys: four 16x16 C tiles, 8 WMMAs -------------
    v8f st[4];
#pragma unroll
    for (int c = 0; c < 4; ++c) {
      const __bf16* kr = Kb + (size_t)(s0 + c * 16 + l16) * DK_;  // key rows
      v8f acc = zero;
      acc = wmma_bf16(qa0, ld_frag(kr + 16 * half, kr + 16 * half + 8), acc);
      acc = wmma_bf16(qa1, ld_frag(kr + 32 + 16 * half, kr + 40 + 16 * half), acc);
      st[c] = acc;
    }

    // --- online softmax in C-layout (one update per 64 keys) --------------
    const bool needMask = (s0 + 63) > q0;   // wave-uniform
#pragma unroll
    for (int r = 0; r < 8; ++r) {
      float v[4];
#pragma unroll
      for (int c = 0; c < 4; ++c) v[c] = st[c][r];
      if (needMask) {
        const int q = q0 + r + 8 * half;
#pragma unroll
        for (int c = 0; c < 4; ++c)
          if (s0 + c * 16 + l16 > q) v[c] = -1e30f;
      }
      float t = fmaxf(fmaxf(v[0], v[1]), fmaxf(v[2], v[3]));
#pragma unroll
      for (int off = 8; off >= 1; off >>= 1)
        t = fmaxf(t, __shfl_xor(t, off, 32));   // row-max over 16-lane group
      const float mn = fmaxf(m[r], t);
      const float sc = __expf(m[r] - mn);
      float p[4];
#pragma unroll
      for (int c = 0; c < 4; ++c) p[c] = __expf(v[c] - mn);
      float u = (p[0] + p[1]) + (p[2] + p[3]);
#pragma unroll
      for (int off = 8; off >= 1; off >>= 1)
        u += __shfl_xor(u, off, 32);            // row-sum over 16-lane group
      l[r] = l[r] * sc + u;
      m[r] = mn;
#pragma unroll
      for (int nt = 0; nt < 4; ++nt) o[nt][r] *= sc;
      const int row = r + 8 * half;
#pragma unroll
      for (int c = 0; c < 4; ++c)
        pbuf[wave][row][c * 16 + l16] = (__bf16)p[c];   // C-layout -> LDS
    }
    // Lanes of a wave are lockstep; only need the DS counter drained before
    // re-reading our own tile in A-layout.
    asm volatile("s_wait_dscnt 0" ::: "memory");

    const __bf16* prow = &pbuf[wave][l16][0];
    v16bf pf0 = ld_frag(prow + 16 * half, prow + 16 * half + 8);        // keys 0..31
    v16bf pf1 = ld_frag(prow + 32 + 16 * half, prow + 40 + 16 * half);   // keys 32..63

    // --- O += P * V : 8 WMMAs (V stored transposed -> contiguous B frags) -
#pragma unroll
    for (int nt = 0; nt < 4; ++nt) {
      const __bf16* vr = Vb + (size_t)(nt * 16 + l16) * T_ + s0 + 16 * half;
      o[nt] = wmma_bf16(pf0, ld_frag(vr, vr + 8), o[nt]);
      const __bf16* vr2 = vr + 32;
      o[nt] = wmma_bf16(pf1, ld_frag(vr2, vr2 + 8), o[nt]);
    }
  }

  // --- normalize and write fp32 output ------------------------------------
  float* ob = out + (size_t)b * T_ * DK_;
#pragma unroll
  for (int r = 0; r < 8; ++r) {
    const float inv = 1.0f / l[r];
    const int q = q0 + r + 8 * half;
#pragma unroll
    for (int nt = 0; nt < 4; ++nt)
      ob[(size_t)q * DK_ + nt * 16 + l16] = o[nt][r] * inv;
  }
}

extern "C" void kernel_launch(void* const* d_in, const int* in_sizes, int n_in,
                              void* d_out, int out_size, void* d_ws, size_t ws_size,
                              hipStream_t stream) {
  const float* X  = (const float*)d_in[0];
  const float* Wq = (const float*)d_in[1];
  const float* bq = (const float*)d_in[2];
  const float* Wk = (const float*)d_in[3];
  const float* bk = (const float*)d_in[4];
  const float* Wv = (const float*)d_in[5];
  const float* bv = (const float*)d_in[6];
  float* out = (float*)d_out;

  __bf16* qws  = (__bf16*)d_ws;                       // [B*T, 64] bf16, pre-scaled
  __bf16* kws  = qws + (size_t)B_ * T_ * DK_;         // [B*T, 64] bf16
  __bf16* vtws = kws + (size_t)B_ * T_ * DK_;         // [B, 64, T] bf16 (transposed)
  __bf16* wtws = vtws + (size_t)B_ * DK_ * T_;        // [3, 64, 1024] bf16 (W^T)

  // 3*64*1024 elements / 256 threads = 768 blocks
  wtrans_kernel<<<dim3(768), dim3(256), 0, stream>>>(Wq, Wk, Wv, wtws);

  // 16384 rows / (16 rows/wave * 4 waves/block) = 256 blocks
  qkv_proj_kernel<<<dim3(256), dim3(128), 0, stream>>>(
      X, wtws, bq, bk, bv, qws, kws, vtws);

  // 32 query-blocks of 64 rows per batch, 8 batches
  attn_kernel<<<dim3(32, 8), dim3(128), 0, stream>>>(qws, kws, vtws, out);
}
